// dec_rnn3_7868380086826
// MI455X (gfx1250) — compile-verified, matrix operand order checked by reading, not run on hardware
//
#include <hip/hip_runtime.h>
#include <hip/hip_fp16.h>
#include <stdint.h>

typedef __attribute__((ext_vector_type(16))) _Float16 v16h;
typedef __attribute__((ext_vector_type(8)))  float    v8f;

#define BB     32
#define QQ     2048
#define KREF   512
#define LAT    20
#define HID    32
#define EMB    16
#define D2H    64
#define OUTD   41

static __device__ __forceinline__ v8f wmma16(v16h a, v16h b, v8f c) {
  // D = A(16x32 f16) * B(32x16 f16) + C(16x16 f32)
  return __builtin_amdgcn_wmma_f32_16x16x32_f16(
      /*neg_a=*/false, a, /*neg_b=*/false, b,
      /*c_mod=*/(short)0, c, /*reuse_a=*/false, /*reuse_b=*/false);
}

// CDNA5 async global->LDS copy (ASYNCcnt-tracked). VDST = per-lane LDS byte
// offset (low 32 bits of generic shared pointer), VADDR = 64-bit global addr.
static __device__ __forceinline__ void async_copy_b128(uint32_t lds_byte,
                                                       const void* gsrc) {
  asm volatile("global_load_async_to_lds_b128 %0, %1, off"
               :: "v"(lds_byte), "v"((uint64_t)(uintptr_t)gsrc) : "memory");
}
static __device__ __forceinline__ void wait_async0() {
  asm volatile("s_wait_asynccnt 0x0" ::: "memory");
}

// ---------------------------------------------------------------------------
// Kernel A: GRU input projection  gi[dir][b][t][96] = z[b][t][:] @ Wih^T + bih
// ---------------------------------------------------------------------------
__global__ void gru_gi_kernel(const float* __restrict__ z,
                              const float* __restrict__ Wih_f, const float* __restrict__ bih_f,
                              const float* __restrict__ Wih_b, const float* __restrict__ bih_b,
                              float* __restrict__ gi) {
  int idx = blockIdx.x * blockDim.x + threadIdx.x;
  const int N = 2 * BB * KREF * 96;
  if (idx >= N) return;
  int row = idx % 96;
  int tmp = idx / 96;
  int t = tmp % KREF; tmp /= KREF;
  int b = tmp % BB;
  int dir = tmp / BB;
  const float* Wih = dir ? Wih_b : Wih_f;
  const float* bih = dir ? bih_b : bih_f;
  const float* zr  = z + ((size_t)b * KREF + t) * LAT;
  const float* wr  = Wih + row * LAT;
  float s = bih[row];
#pragma unroll
  for (int i = 0; i < LAT; ++i) s += zr[i] * wr[i];
  gi[idx] = s;
}

// ---------------------------------------------------------------------------
// Kernel B: GRU recurrence. 64 blocks = (batch, direction), one wave each.
// ---------------------------------------------------------------------------
__global__ void __launch_bounds__(32)
gru_rec_kernel(const float* __restrict__ gi,
               const float* __restrict__ Whh_f, const float* __restrict__ bhh_f,
               const float* __restrict__ Whh_b, const float* __restrict__ bhh_b,
               float* __restrict__ val) {
  const int b   = blockIdx.x >> 1;
  const int dir = blockIdx.x & 1;
  const int lane = threadIdx.x;
  const float* Whh = dir ? Whh_b : Whh_f;
  const float* bhh = dir ? bhh_b : bhh_f;

  float wr[HID], wz[HID], wn[HID];
#pragma unroll
  for (int c = 0; c < HID; ++c) {
    wr[c] = Whh[(0 * HID + lane) * HID + c];
    wz[c] = Whh[(1 * HID + lane) * HID + c];
    wn[c] = Whh[(2 * HID + lane) * HID + c];
  }
  const float br = bhh[lane], bz = bhh[HID + lane], bn = bhh[2 * HID + lane];

  __shared__ float hs[HID];
  float h = 0.0f;
  hs[lane] = 0.0f;
  __syncthreads();

  const float* gbase = gi + (((size_t)dir * BB + b) * KREF) * 96;
  for (int step = 0; step < KREF; ++step) {
    const int t = dir ? (KREF - 1 - step) : step;
    const float* g = gbase + (size_t)t * 96;
    const float gr = g[lane], gz = g[HID + lane], gn = g[2 * HID + lane];
    float ar = br, az = bz, an = bn;
#pragma unroll
    for (int c = 0; c < HID; ++c) {
      const float hc = hs[c];
      ar += wr[c] * hc; az += wz[c] * hc; an += wn[c] * hc;
    }
    const float r = 1.0f / (1.0f + __expf(-(gr + ar)));
    const float u = 1.0f / (1.0f + __expf(-(gz + az)));
    const float n = tanhf(gn + r * an);
    h = (1.0f - u) * n + u * h;
    __syncthreads();
    hs[lane] = h;
    __syncthreads();
    val[((size_t)b * KREF + t) * D2H + dir * HID + lane] = h;
  }
}

// ---------------------------------------------------------------------------
// Kernel C1/C2: time embedding + projection -> f16 (query and key)
// ---------------------------------------------------------------------------
__device__ __forceinline__ void time_embed16(float pos, float* e) {
#pragma unroll
  for (int j = 0; j < 8; ++j) {
    const float div = __expf(-0.28782313662425572f * (float)j); // -2j*ln(10)/16
    const float arg = 48.0f * pos * div;
    e[2 * j]     = __sinf(arg);
    e[2 * j + 1] = __cosf(arg);
  }
}

__global__ void q_embed_kernel(const float* __restrict__ ts,
                               const float* __restrict__ Wq, const float* __restrict__ bq,
                               _Float16* __restrict__ qh) {
  int idx = blockIdx.x * blockDim.x + threadIdx.x;
  if (idx >= BB * QQ) return;
  float e[EMB];
  time_embed16(ts[idx], e);
  _Float16* o = qh + (size_t)idx * EMB;
#pragma unroll
  for (int r = 0; r < EMB; ++r) {
    const float* w = Wq + r * EMB;
    float s = bq[r];
#pragma unroll
    for (int c = 0; c < EMB; ++c) s += w[c] * e[c];
    o[r] = (_Float16)s;
  }
}

__global__ void k_embed_kernel(const float* __restrict__ rq,
                               const float* __restrict__ Wk, const float* __restrict__ bk,
                               _Float16* __restrict__ kh) {
  int idx = blockIdx.x * blockDim.x + threadIdx.x;
  if (idx >= KREF) return;
  float e[EMB];
  time_embed16(rq[idx], e);
  _Float16* o = kh + (size_t)idx * EMB;
#pragma unroll
  for (int r = 0; r < EMB; ++r) {
    const float* w = Wk + r * EMB;
    float s = bk[r];
#pragma unroll
    for (int c = 0; c < EMB; ++c) s += w[c] * e[c];
    o[r] = (_Float16)s;
  }
}

// ---------------------------------------------------------------------------
// Kernel C3: v projection, transposed to [B][64][KREF] f16 for WMMA B-frags
// ---------------------------------------------------------------------------
__global__ void v_proj_kernel(const float* __restrict__ val,
                              const float* __restrict__ Wv, const float* __restrict__ bv,
                              _Float16* __restrict__ vth) {
  int idx = blockIdx.x * blockDim.x + threadIdx.x;
  if (idx >= BB * D2H * KREF) return;
  int k = idx % KREF;
  int tmp = idx / KREF;
  int d = tmp % D2H;
  int b = tmp / D2H;
  const float* vr = val + ((size_t)b * KREF + k) * D2H;
  const float* w  = Wv + d * D2H;
  float s = bv[d];
#pragma unroll
  for (int c = 0; c < D2H; ++c) s += vr[c] * w[c];
  vth[idx] = (_Float16)s;
}

// ---------------------------------------------------------------------------
// Kernel D: fused flash-attention (WMMA) + MLP.
// 4 waves/block, all on the same batch; K staged once in LDS, V tiles
// double-buffered through LDS with async global->LDS copies (ASYNCcnt).
// ---------------------------------------------------------------------------
__global__ void __launch_bounds__(128)
attn_mlp_kernel(const _Float16* __restrict__ qh, const _Float16* __restrict__ khg,
                const _Float16* __restrict__ vth,
                const float* __restrict__ W1, const float* __restrict__ b1,
                const float* __restrict__ W2, const float* __restrict__ b2,
                float* __restrict__ out) {
  __shared__ _Float16 khs[KREF * EMB];      // 16 KB: all keys, f16
  __shared__ _Float16 vbuf[2][D2H * 32];    // 2 x 4 KB: double-buffered V tile
  __shared__ float pbuf[4][16 * 32];        // per-wave P re-layout slab
  __shared__ float attb[4][16 * 64];        // per-wave att tile
  __shared__ float hb[4][16 * 50];          // per-wave MLP hidden

  const int tid  = threadIdx.x;
  const int lane = tid & 31;
  const int w    = tid >> 5;
  const int tile = blockIdx.x * 4 + w;
  const int b    = tile >> 7;               // 128 tiles per batch, 4/block
  const int q0   = (tile & 127) * 16;

  const int m16 = lane & 15;
  const int hi  = lane >> 4;
  const int kb  = hi * 8;

  const _Float16* vbb = vth + (size_t)b * D2H * KREF;

  // ---- async-stage all keys (16 KB, contiguous) ----
  {
    const char* ksrc = (const char*)khg + tid * 128;
    const uint32_t kdst = (uint32_t)(uintptr_t)khs + tid * 128;
#pragma unroll
    for (int i = 0; i < 8; ++i) async_copy_b128(kdst + i * 16, ksrc + i * 16);
  }
  // V tile loader: 64 rows x 64 B; thread covers row tid>>1, 32 B half tid&1
  auto load_vtile = [&](int k0, int bsel) {
    const int d = tid >> 1, part = tid & 1;
    const char* src = (const char*)(vbb + (size_t)d * KREF + k0) + part * 32;
    const uint32_t dst =
        (uint32_t)(uintptr_t)&vbuf[bsel][d * 32] + part * 32;
    async_copy_b128(dst, src);
    async_copy_b128(dst + 16, src + 16);
  };
  load_vtile(0, 0);
  wait_async0();
  __syncthreads();

  // ---- A fragment: 16 queries x (E=16 padded to K=32) ----
  v16h aq = {};
  {
    const _Float16* qrow = qh + ((size_t)(b * QQ + q0 + m16)) * EMB + kb;
#pragma unroll
    for (int ev = 0; ev < 8; ++ev) aq[ev] = qrow[ev];   // regs 4-7 zero (pad)
  }

  v8f acc0 = {}, acc1 = {}, acc2 = {}, acc3 = {};
  float mrow[8], lrow[8];
#pragma unroll
  for (int r = 0; r < 8; ++r) { mrow[r] = -1e30f; lrow[r] = 0.0f; }

  for (int c = 0; c < KREF / 32; ++c) {
    const int k0 = c * 32;
    if (c + 1 < KREF / 32) load_vtile(k0 + 32, (c + 1) & 1);
    const _Float16* vs = &vbuf[c & 1][0];

    // ---- key B-fragments from LDS (k^T, E padded to 32) ----
    v16h bk0v = {}, bk1v = {};
    {
      const _Float16* kr0 = khs + (size_t)(k0 + m16) * EMB + kb;
      const _Float16* kr1 = khs + (size_t)(k0 + 16 + m16) * EMB + kb;
#pragma unroll
      for (int ev = 0; ev < 8; ++ev) { bk0v[ev] = kr0[ev]; bk1v[ev] = kr1[ev]; }
    }
    v8f z0 = {}, z1 = {};
    v8f s0 = wmma16(aq, bk0v, z0);
    v8f s1 = wmma16(aq, bk1v, z1);

    // ---- online softmax over this 32-key chunk ----
    float alpha[8], p0a[8], p1a[8];
#pragma unroll
    for (int r = 0; r < 8; ++r) {
      const float sa = s0[r] * 0.25f;      // 1/sqrt(16)
      const float sb = s1[r] * 0.25f;
      float t = fmaxf(sa, sb);
#pragma unroll
      for (int off = 1; off < 16; off <<= 1) t = fmaxf(t, __shfl_xor(t, off, 16));
      const float mnew = fmaxf(mrow[r], t);
      alpha[r] = __expf(mrow[r] - mnew);
      mrow[r] = mnew;
      const float pa = __expf(sa - mnew);
      const float pb = __expf(sb - mnew);
      p0a[r] = pa; p1a[r] = pb;
      float rs = pa + pb;
#pragma unroll
      for (int off = 1; off < 16; off <<= 1) rs += __shfl_xor(rs, off, 16);
      lrow[r] = lrow[r] * alpha[r] + rs;
      acc0[r] *= alpha[r]; acc1[r] *= alpha[r];
      acc2[r] *= alpha[r]; acc3[r] *= alpha[r];
    }

    // ---- bounce P through per-wave LDS slab into an A fragment ----
    float* pb_ = pbuf[w];
#pragma unroll
    for (int r = 0; r < 8; ++r) {
      pb_[(r + 8 * hi) * 32 + m16]      = p0a[r];
      pb_[(r + 8 * hi) * 32 + 16 + m16] = p1a[r];
    }
    v16h ap;
    {
      const float* src = pb_ + m16 * 32;
#pragma unroll
      for (int ev = 0; ev < 8; ++ev) ap[ev]     = (_Float16)src[kb + ev];
#pragma unroll
      for (int ev = 0; ev < 8; ++ev) ap[8 + ev] = (_Float16)src[16 + kb + ev];
    }

    // ---- V B-fragments from the LDS tile: rows d, 32 keys each ----
    v16h bv0, bv1, bv2, bv3;
    {
      const _Float16* vr0 = vs + (size_t)(0 * 16 + m16) * 32 + kb;
      const _Float16* vr1 = vs + (size_t)(1 * 16 + m16) * 32 + kb;
      const _Float16* vr2 = vs + (size_t)(2 * 16 + m16) * 32 + kb;
      const _Float16* vr3 = vs + (size_t)(3 * 16 + m16) * 32 + kb;
#pragma unroll
      for (int ev = 0; ev < 8; ++ev) {
        bv0[ev] = vr0[ev]; bv0[8 + ev] = vr0[16 + ev];
        bv1[ev] = vr1[ev]; bv1[8 + ev] = vr1[16 + ev];
        bv2[ev] = vr2[ev]; bv2[8 + ev] = vr2[16 + ev];
        bv3[ev] = vr3[ev]; bv3[8 + ev] = vr3[16 + ev];
      }
    }
    acc0 = wmma16(ap, bv0, acc0);
    acc1 = wmma16(ap, bv1, acc1);
    acc2 = wmma16(ap, bv2, acc2);
    acc3 = wmma16(ap, bv3, acc3);

    if (c + 1 < KREF / 32) wait_async0();   // next tile resident in LDS
    __syncthreads();                        // publish tile / retire reads
  }

  // ---- normalize and stage att tile (16x64 f32) in LDS ----
  float* ab = attb[w];
#pragma unroll
  for (int r = 0; r < 8; ++r) {
    const float inv = 1.0f / lrow[r];
    const int row = (r + 8 * hi) * 64;
    ab[row + 0 * 16 + m16] = acc0[r] * inv;
    ab[row + 1 * 16 + m16] = acc1[r] * inv;
    ab[row + 2 * 16 + m16] = acc2[r] * inv;
    ab[row + 3 * 16 + m16] = acc3[r] * inv;
  }
  __syncthreads();

  // ---- MLP: 64 -> relu(50) -> 41, two lanes per query ----
  const int qm = lane >> 1;
  const int half = lane & 1;
  const float* arow = attb[w] + qm * 64;
  float* hrow = hb[w] + qm * 50;
#pragma unroll
  for (int j = 0; j < 25; ++j) {
    const int jj = half * 25 + j;
    const float* w1r = W1 + jj * 64;
    float s = b1[jj];
#pragma unroll
    for (int c2 = 0; c2 < D2H; ++c2) s += arow[c2] * w1r[c2];
    hrow[jj] = fmaxf(s, 0.0f);
  }
  __syncthreads();

  float* orow = out + ((size_t)(b * QQ + q0 + qm)) * OUTD;
  for (int j = half; j < OUTD; j += 2) {
    const float* w2r = W2 + j * 50;
    float s = b2[j];
#pragma unroll
    for (int c2 = 0; c2 < 50; ++c2) s += hrow[c2] * w2r[c2];
    orow[j] = s;
  }
}

// ---------------------------------------------------------------------------
extern "C" void kernel_launch(void* const* d_in, const int* in_sizes, int n_in,
                              void* d_out, int out_size, void* d_ws, size_t ws_size,
                              hipStream_t stream) {
  const float* z     = (const float*)d_in[0];
  const float* ts    = (const float*)d_in[1];
  const float* rq    = (const float*)d_in[2];
  const float* Wih_f = (const float*)d_in[3];
  const float* Whh_f = (const float*)d_in[4];
  const float* bih_f = (const float*)d_in[5];
  const float* bhh_f = (const float*)d_in[6];
  const float* Wih_b = (const float*)d_in[7];
  const float* Whh_b = (const float*)d_in[8];
  const float* bih_b = (const float*)d_in[9];
  const float* bhh_b = (const float*)d_in[10];
  const float* Wq    = (const float*)d_in[11];
  const float* bq    = (const float*)d_in[12];
  const float* Wk    = (const float*)d_in[13];
  const float* bk    = (const float*)d_in[14];
  const float* Wv    = (const float*)d_in[15];
  const float* bv    = (const float*)d_in[16];
  const float* W1    = (const float*)d_in[17];
  const float* b1    = (const float*)d_in[18];
  const float* W2    = (const float*)d_in[19];
  const float* b2    = (const float*)d_in[20];

  // workspace layout
  float* gi  = (float*)d_ws;                          // 2*B*K*96
  float* val = gi + (size_t)2 * BB * KREF * 96;       // B*K*64
  _Float16* qh  = (_Float16*)(val + (size_t)BB * KREF * D2H);  // B*Q*16
  _Float16* kh  = qh + (size_t)BB * QQ * EMB;                  // K*16
  _Float16* vth = kh + (size_t)KREF * EMB;                     // B*64*K

  {
    const int N = 2 * BB * KREF * 96;
    gru_gi_kernel<<<(N + 255) / 256, 256, 0, stream>>>(z, Wih_f, bih_f, Wih_b, bih_b, gi);
  }
  gru_rec_kernel<<<2 * BB, 32, 0, stream>>>(gi, Whh_f, bhh_f, Whh_b, bhh_b, val);
  {
    const int N = BB * QQ;
    q_embed_kernel<<<(N + 255) / 256, 256, 0, stream>>>(ts, Wq, bq, qh);
  }
  k_embed_kernel<<<(KREF + 255) / 256, 256, 0, stream>>>(rq, Wk, bk, kh);
  {
    const int N = BB * D2H * KREF;
    v_proj_kernel<<<(N + 255) / 256, 256, 0, stream>>>(val, Wv, bv, vth);
  }
  {
    const int tiles = BB * (QQ / 16);   // 4096 waves, 4 per block
    attn_mlp_kernel<<<tiles / 4, 128, 0, stream>>>(qh, kh, vth, W1, b1, W2, b2,
                                                   (float*)d_out);
  }
}
